// ScaledDotProductAttention_56118042689668
// MI455X (gfx1250) — matrix operand order repaired and checked by reading, not compiled, not run on hardware
//
#include <hip/hip_runtime.h>

// ============================================================================
// Scaled-dot-product attention (B=8, T=2048, D=1024, single head) for gfx1250.
//
// Pipeline:
//   k_qkv   : q = (x@Wq+bq)*D^-0.5, k = x@Wk+bk  (row-major bf16)
//             v = x@Wv+bv stored TRANSPOSED as vt[b][d][t] (bf16)
//   k_stats : per-row softmax max m and sum-exp l over S = q@k^T (WMMA bf16)
//   k_attn  : O = (exp(S-m)/l) @ V, S recomputed per d-half (WMMA bf16)
//
// CDNA5 specifics used:
//   * v_wmma_f32_16x16x32_bf16 (f32 accumulate) for all matmuls
//   * global_load_async_to_lds_b128 (ASYNCcnt) double-buffered staging with a
//     single workgroup barrier per K-step (guarded by __has_builtin; falls
//     back to synchronous b128 copies)
//
// WMMA fragment conventions used here (wave32; A/B k order chosen by this code, same in both):
//   A (16x32 bf16): lane l (g=l/16, m=l%16) holds halves e with
//                   K = 16*g + e                -> two contiguous 16B loads.
//   B (32x16 bf16): lane l (g=l/16, n=l%16) holds halves e with
//                   K = 16*g + e                -> two contiguous 16B loads
//                   from a K-contiguous (N-major) buffer.
//   C/D (16x16 f32): VGPR r, lanes 0-15: M=r, N=lane; lanes 16-31: M=r+8.
// ============================================================================

typedef __bf16 bf16_t;
typedef bf16_t v8bf  __attribute__((ext_vector_type(8)));
typedef bf16_t v16bf __attribute__((ext_vector_type(16)));
typedef float  v8f   __attribute__((ext_vector_type(8)));

#if defined(__has_builtin)
#  if __has_builtin(__builtin_amdgcn_global_load_async_to_lds_b128) && \
      __has_builtin(__builtin_amdgcn_s_wait_asynccnt)
#    define HAVE_ASYNC 1
#  endif
#endif
#ifndef HAVE_ASYNC
#  define HAVE_ASYNC 0
#endif

// 16-byte global -> LDS copy (per lane). Async path uses the CDNA5 direct
// global->LDS DMA tracked by ASYNCcnt; fallback is a VGPR bounce.
// Builtin prototype (from hipcc diagnostic): arg1 = vector_size(16) int
// pointer in the global (__device__/AS1) address space.
static __device__ __forceinline__ void cp16_g2l(const void* g, void* l) {
#if HAVE_ASYNC
  typedef int v4i_ __attribute__((vector_size(16)));
  __builtin_amdgcn_global_load_async_to_lds_b128(
      (__attribute__((address_space(1))) v4i_*)(v4i_*)g,
      (__attribute__((address_space(3))) v4i_*)(v4i_*)l, 0, 0);
#else
  *(uint4*)l = *(const uint4*)g;
#endif
}

static __device__ __forceinline__ void async_wait() {
#if HAVE_ASYNC
  __builtin_amdgcn_s_wait_asynccnt(0);
#endif
}

static __device__ __forceinline__ unsigned short f2bf(float f) {
  union { float f; unsigned u; } cv; cv.f = f;
  unsigned u = cv.u;
  u += 0x7fffu + ((u >> 16) & 1u);            // round-to-nearest-even
  return (unsigned short)(u >> 16);
}

static __device__ __forceinline__ unsigned pk2(float a, float b) {
  return (unsigned)f2bf(a) | ((unsigned)f2bf(b) << 16);
}

static __device__ __forceinline__ v16bf cat16(v8bf lo, v8bf hi) {
  return __builtin_shufflevector(lo, hi, 0,1,2,3,4,5,6,7,8,9,10,11,12,13,14,15);
}

static __device__ __forceinline__ v8f wmma_bf16(v16bf a, v16bf b, v8f c) {
  return __builtin_amdgcn_wmma_f32_16x16x32_bf16(
      /*neg_a=*/false, a, /*neg_b=*/false, b,
      /*c_mod=*/(short)0, c, /*reuse_a=*/false, /*reuse_b=*/false);
}

// A fragment: rows along M, halves K-contiguous in base[row*stride + k].
static __device__ __forceinline__ v16bf load_a_frag(const unsigned short* base,
                                                    int row, int stride, int g) {
  const v8bf* p0 = (const v8bf*)(base + row * stride + 16 * g);
  const v8bf* p1 = (const v8bf*)(base + row * stride + 16 * g + 8);
  return cat16(*p0, *p1);
}

// B fragment: "rows" along N, halves K-contiguous in base[n*stride + k].
static __device__ __forceinline__ v16bf load_b_frag(const unsigned short* base,
                                                    int n, int stride, int g) {
  const v8bf* p0 = (const v8bf*)(base + n * stride + 16 * g);
  const v8bf* p1 = (const v8bf*)(base + n * stride + 16 * g + 8);
  return cat16(*p0, *p1);
}

#define ZERO8 {0.f,0.f,0.f,0.f,0.f,0.f,0.f,0.f}

// ============================================================================
// Kernel 1: QKV projections. grid (128 Mtiles, 8 Ntiles, 3 matrices), 256 thr.
// Block tile 128x128, 8 waves in 2x4, wave tile 64x32 (4x2 WMMA accums).
// (Synchronous staging: converts f32 -> bf16 in flight.)
// ============================================================================
__global__ __launch_bounds__(256) void k_qkv(
    const float* __restrict__ x,
    const float* __restrict__ Wq, const float* __restrict__ bq,
    const float* __restrict__ Wk, const float* __restrict__ bk,
    const float* __restrict__ Wv, const float* __restrict__ bv,
    unsigned short* __restrict__ qb, unsigned short* __restrict__ kb,
    unsigned short* __restrict__ vt) {
  const int z = blockIdx.z;
  const float* W    = (z == 0) ? Wq : (z == 1) ? Wk : Wv;
  const float* bias = (z == 0) ? bq : (z == 1) ? bk : bv;
  const int mBase = blockIdx.x * 128;
  const int nBase = blockIdx.y * 128;

  __shared__ __align__(16) unsigned short lA[128 * 32];   // x tile, M-major
  __shared__ __align__(16) unsigned short lB[128 * 32];   // W tile, N-major

  const int tid  = threadIdx.x;
  const int lane = tid & 31;
  const int wave = tid >> 5;
  const int wm = wave >> 2;          // 0..1
  const int wn = wave & 3;           // 0..3
  const int g  = lane >> 4;
  const int ln = lane & 15;

  v8f acc[4][2];
#pragma unroll
  for (int i = 0; i < 4; i++)
#pragma unroll
    for (int j = 0; j < 2; j++) acc[i][j] = (v8f)ZERO8;

  for (int k0 = 0; k0 < 1024; k0 += 32) {
    {   // stage A: x[128 x 32] f32 -> bf16, M-major
      const int row = tid >> 1;
      const int kk  = (tid & 1) * 16;
      const float4* src = (const float4*)(x + (size_t)(mBase + row) * 1024 + k0 + kk);
      float4 f0 = src[0], f1 = src[1], f2 = src[2], f3 = src[3];
      uint4 d0 = make_uint4(pk2(f0.x, f0.y), pk2(f0.z, f0.w),
                            pk2(f1.x, f1.y), pk2(f1.z, f1.w));
      uint4 d1 = make_uint4(pk2(f2.x, f2.y), pk2(f2.z, f2.w),
                            pk2(f3.x, f3.y), pk2(f3.z, f3.w));
      uint4* dst = (uint4*)(lA + row * 32 + kk);
      dst[0] = d0; dst[1] = d1;
    }
    {   // stage B: W[32 x 128] f32 -> bf16, stored transposed (N-major)
      const int kr = tid >> 3;              // 0..31
      const int nn = (tid & 7) * 16;        // 0..112
      const float4* src = (const float4*)(W + (size_t)(k0 + kr) * 1024 + nBase + nn);
#pragma unroll
      for (int v = 0; v < 4; v++) {
        float4 f = src[v];
        lB[(nn + 4 * v + 0) * 32 + kr] = f2bf(f.x);
        lB[(nn + 4 * v + 1) * 32 + kr] = f2bf(f.y);
        lB[(nn + 4 * v + 2) * 32 + kr] = f2bf(f.z);
        lB[(nn + 4 * v + 3) * 32 + kr] = f2bf(f.w);
      }
    }
    __syncthreads();
    v16bf af[4], bfr[2];
#pragma unroll
    for (int mt = 0; mt < 4; mt++) af[mt]  = load_a_frag(lA, wm * 64 + mt * 16 + ln, 32, g);
#pragma unroll
    for (int nt = 0; nt < 2; nt++) bfr[nt] = load_b_frag(lB, wn * 32 + nt * 16 + ln, 32, g);
#pragma unroll
    for (int mt = 0; mt < 4; mt++)
#pragma unroll
      for (int nt = 0; nt < 2; nt++)
        acc[mt][nt] = wmma_bf16(af[mt], bfr[nt], acc[mt][nt]);
    __syncthreads();
  }

  const float scale = (z == 0) ? 0.03125f : 1.0f;   // D^-0.5 = 1/32 for q only
#pragma unroll
  for (int nt = 0; nt < 2; nt++) {
    const int col = nBase + wn * 32 + nt * 16 + ln;
    const float bv_ = bias[col];
#pragma unroll
    for (int mt = 0; mt < 4; mt++) {
#pragma unroll
      for (int r = 0; r < 8; r++) {
        const int mrow = mBase + wm * 64 + mt * 16 + r + 8 * g;
        const unsigned short h = f2bf((acc[mt][nt][r] + bv_) * scale);
        if (z == 0) {
          qb[(size_t)mrow * 1024 + col] = h;
        } else if (z == 1) {
          kb[(size_t)mrow * 1024 + col] = h;
        } else {
          const int bb = mrow >> 11, tt = mrow & 2047;     // batch, token
          vt[((size_t)bb * 1024 + col) * 2048 + tt] = h;   // transposed store
        }
      }
    }
  }
}

// ============================================================================
// Kernel 2: softmax row stats (max m, sum-exp l). grid (16 qtiles, 8 batches),
// 256 thr; wave w owns 16 query rows, S chunks of 16x128 in registers.
// Double-buffered async global->LDS staging, one barrier per K-step.
// ============================================================================
__global__ __launch_bounds__(256) void k_stats(
    const unsigned short* __restrict__ qb, const unsigned short* __restrict__ kb,
    float* __restrict__ mOut, float* __restrict__ lOut) {
  const int b  = blockIdx.y;
  const int q0 = blockIdx.x * 128;
  __shared__ __align__(16) unsigned short lQ[2][128 * 32];
  __shared__ __align__(16) unsigned short lK[2][128 * 32];
  const int tid = threadIdx.x, lane = tid & 31, wave = tid >> 5;
  const int g = lane >> 4, ln = lane & 15;
  const int srow = tid >> 1, skk = (tid & 1) * 16;   // staging: 16 halves/thread

  float rm[8], rl[8];
#pragma unroll
  for (int r = 0; r < 8; r++) { rm[r] = -3.0e38f; rl[r] = 0.0f; }

  const size_t qrow0 = (size_t)(b * 2048 + q0);

  auto issueQK = [&](int buf, int d0, int kc) {
    const unsigned short* gq = qb + (qrow0 + srow) * 1024 + d0 + skk;
    const unsigned short* gk = kb + (size_t)(b * 2048 + kc + srow) * 1024 + d0 + skk;
    unsigned short* sq = &lQ[buf][srow * 32 + skk];
    unsigned short* sk = &lK[buf][srow * 32 + skk];
    cp16_g2l(gq, sq); cp16_g2l(gq + 8, sq + 8);
    cp16_g2l(gk, sk); cp16_g2l(gk + 8, sk + 8);
  };

  for (int kc = 0; kc < 2048; kc += 128) {
    v8f S[8];
#pragma unroll
    for (int nt = 0; nt < 8; nt++) S[nt] = (v8f)ZERO8;

    issueQK(0, 0, kc);                              // prologue
    for (int d0 = 0; d0 < 1024; d0 += 32) {
      const int cur = (d0 >> 5) & 1;
      async_wait();                                 // my copies for `cur` done
      __syncthreads();                              // everyone's copies done
      if (d0 + 32 < 1024) issueQK(cur ^ 1, d0 + 32, kc);
      const v16bf a = load_a_frag(lQ[cur], wave * 16 + ln, 32, g);
#pragma unroll
      for (int nt = 0; nt < 8; nt++) {
        const v16bf bfr = load_b_frag(lK[cur], nt * 16 + ln, 32, g);
        S[nt] = wmma_bf16(a, bfr, S[nt]);
      }
      // no trailing barrier: buffer `cur` is only rewritten after the next
      // iteration's barrier, and ds reads are fenced by the WMMA data deps.
    }
    // online update of (m, l) per row; row r+8g is spread over 16 lanes (cols)
#pragma unroll
    for (int r = 0; r < 8; r++) {
      float tmax = -3.0e38f;
#pragma unroll
      for (int nt = 0; nt < 8; nt++) tmax = fmaxf(tmax, S[nt][r]);
#pragma unroll
      for (int mask = 1; mask < 16; mask <<= 1)
        tmax = fmaxf(tmax, __shfl_xor(tmax, mask, 32));
      const float nm = fmaxf(rm[r], tmax);
      float es = 0.f;
#pragma unroll
      for (int nt = 0; nt < 8; nt++) es += __expf(S[nt][r] - nm);
#pragma unroll
      for (int mask = 1; mask < 16; mask <<= 1)
        es += __shfl_xor(es, mask, 32);
      rl[r] = rl[r] * __expf(rm[r] - nm) + es;
      rm[r] = nm;
    }
  }
  if (ln == 0) {
#pragma unroll
    for (int r = 0; r < 8; r++) {
      const int tok = q0 + wave * 16 + r + 8 * g;
      mOut[b * 2048 + tok] = rm[r];
      lOut[b * 2048 + tok] = rl[r];
    }
  }
}

// ============================================================================
// Kernel 3: O = P @ V. grid (16 qtiles, 8 batches, 2 d-halves), 512 thr
// (16 waves in 4x4). Wave: S region 32x32, O region 32x128 (2x8 accums).
// S-phase uses double-buffered async staging like k_stats.
// ============================================================================
__global__ __launch_bounds__(512) void k_attn(
    const unsigned short* __restrict__ qb, const unsigned short* __restrict__ kb,
    const unsigned short* __restrict__ vt, const float* __restrict__ mIn,
    const float* __restrict__ lIn, float* __restrict__ out) {
  const int b  = blockIdx.y;
  const int q0 = blockIdx.x * 128;
  const int dh = blockIdx.z;                 // d half: 0 or 1
  __shared__ __align__(16) unsigned short lQ[2][128 * 32];
  __shared__ __align__(16) unsigned short lK[2][128 * 32];
  __shared__ __align__(16) unsigned short lP[128 * 128];
  __shared__ float lM[128];
  __shared__ float lL[128];                  // holds 1/l

  const int tid = threadIdx.x, lane = tid & 31, wave = tid >> 5;
  const int wr = wave >> 2, wc = wave & 3;
  const int g = lane >> 4, ln = lane & 15;
  const int srow = tid >> 2, skk = (tid & 3) * 8;   // staging: 8 halves/thread

  if (tid < 128) {
    lM[tid] = mIn[b * 2048 + q0 + tid];
    lL[tid] = 1.0f / lIn[b * 2048 + q0 + tid];
  }
  __syncthreads();

  v8f O[2][8];
#pragma unroll
  for (int mt = 0; mt < 2; mt++)
#pragma unroll
    for (int dt = 0; dt < 8; dt++) O[mt][dt] = (v8f)ZERO8;

  const int dBase = dh * 512 + wc * 128;
  const size_t qrow0 = (size_t)(b * 2048 + q0);

  auto issueQK = [&](int buf, int d0, int kc) {
    cp16_g2l(qb + (qrow0 + srow) * 1024 + d0 + skk, &lQ[buf][srow * 32 + skk]);
    cp16_g2l(kb + (size_t)(b * 2048 + kc + srow) * 1024 + d0 + skk,
             &lK[buf][srow * 32 + skk]);
  };

  for (int kc = 0; kc < 2048; kc += 128) {
    v8f S[2][2];
#pragma unroll
    for (int mt = 0; mt < 2; mt++)
#pragma unroll
      for (int nt = 0; nt < 2; nt++) S[mt][nt] = (v8f)ZERO8;

    issueQK(0, 0, kc);                              // prologue
    for (int d0 = 0; d0 < 1024; d0 += 32) {
      const int cur = (d0 >> 5) & 1;
      async_wait();
      __syncthreads();
      if (d0 + 32 < 1024) issueQK(cur ^ 1, d0 + 32, kc);
      const v16bf a0 = load_a_frag(lQ[cur], wr * 32 + ln, 32, g);
      const v16bf a1 = load_a_frag(lQ[cur], wr * 32 + 16 + ln, 32, g);
      const v16bf b0 = load_b_frag(lK[cur], wc * 32 + ln, 32, g);
      const v16bf b1 = load_b_frag(lK[cur], wc * 32 + 16 + ln, 32, g);
      S[0][0] = wmma_bf16(a0, b0, S[0][0]);
      S[0][1] = wmma_bf16(a0, b1, S[0][1]);
      S[1][0] = wmma_bf16(a1, b0, S[1][0]);
      S[1][1] = wmma_bf16(a1, b1, S[1][1]);
    }
    __syncthreads();   // S reads of lQ/lK done block-wide before lP phase races ahead
    // P = exp(S - m) * (1/l), stored bf16 in lP (row-major over keys)
#pragma unroll
    for (int mt = 0; mt < 2; mt++) {
#pragma unroll
      for (int nt = 0; nt < 2; nt++) {
        const int col = wc * 32 + nt * 16 + ln;
#pragma unroll
        for (int r = 0; r < 8; r++) {
          const int row = wr * 32 + mt * 16 + r + 8 * g;
          const float p = __expf(S[mt][nt][r] - lM[row]) * lL[row];
          lP[row * 128 + col] = f2bf(p);
        }
      }
    }
    __syncthreads();
    // O += P(128x128) @ V(128x512-half); V streamed from transposed vt (L2)
#pragma unroll
    for (int kk = 0; kk < 128; kk += 32) {
      const v16bf pa0 = load_a_frag(lP + kk, wr * 32 + ln, 128, g);
      const v16bf pa1 = load_a_frag(lP + kk, wr * 32 + 16 + ln, 128, g);
#pragma unroll
      for (int dt = 0; dt < 8; dt++) {
        const int dcol = dBase + dt * 16 + ln;
        const unsigned short* vp =
            vt + (size_t)(b * 1024 + dcol) * 2048 + kc + kk + 16 * g;
        const v16bf bv16 = cat16(*(const v8bf*)vp, *(const v8bf*)(vp + 8));
        O[0][dt] = wmma_bf16(pa0, bv16, O[0][dt]);
        O[1][dt] = wmma_bf16(pa1, bv16, O[1][dt]);
      }
    }
    __syncthreads();   // all lP reads done before next chunk rewrites it
  }

#pragma unroll
  for (int dt = 0; dt < 8; dt++) {
    const int dcol = dBase + dt * 16 + ln;
#pragma unroll
    for (int mt = 0; mt < 2; mt++) {
#pragma unroll
      for (int r = 0; r < 8; r++) {
        const int tok = q0 + wr * 32 + mt * 16 + r + 8 * g;
        out[(size_t)(b * 2048 + tok) * 1024 + dcol] = O[mt][dt][r];
      }
    }
  }
}

// ============================================================================
extern "C" void kernel_launch(void* const* d_in, const int* in_sizes, int n_in,
                              void* d_out, int out_size, void* d_ws, size_t ws_size,
                              hipStream_t stream) {
  (void)in_sizes; (void)n_in; (void)out_size; (void)ws_size;
  const float* x  = (const float*)d_in[0];
  const float* Wq = (const float*)d_in[1];
  const float* bq = (const float*)d_in[2];
  const float* Wk = (const float*)d_in[3];
  const float* bk = (const float*)d_in[4];
  const float* Wv = (const float*)d_in[5];
  const float* bv = (const float*)d_in[6];
  float* out = (float*)d_out;

  // ws layout: 3 x 16777216 bf16 (qb, kb, vt) + 2 x 16384 f32 stats (~96 MiB)
  unsigned short* qb = (unsigned short*)d_ws;
  unsigned short* kb = qb + (size_t)16777216;
  unsigned short* vt = kb + (size_t)16777216;
  float* mstat = (float*)(vt + (size_t)16777216);
  float* lstat = mstat + 16384;

  k_qkv <<<dim3(128, 8, 3), dim3(256), 0, stream>>>(x, Wq, bq, Wk, bk, Wv, bv,
                                                    qb, kb, vt);
  k_stats<<<dim3(16, 8),     dim3(256), 0, stream>>>(qb, kb, mstat, lstat);
  k_attn <<<dim3(16, 8, 2),  dim3(512), 0, stream>>>(qb, kb, vt, mstat, lstat, out);
}
